// GraphAnalystAgent_34153579938349
// MI455X (gfx1250) — matrix operand order repaired and checked by reading, not compiled, not run on hardware
//
#include <hip/hip_runtime.h>
#include <hip/hip_bf16.h>
#include <math.h>

#define NNODES_  16384
#define INDIM    768
#define HID1     256
#define H1N      4
#define OUTC     128
#define H2N      2
#define NGRAPH   8

typedef __attribute__((ext_vector_type(16))) __bf16 v16bf;
typedef __attribute__((ext_vector_type(8)))  __bf16 v8bf;
typedef __attribute__((ext_vector_type(8)))  float  v8f;

__device__ __forceinline__ float lrelu02(float x){ return x > 0.f ? x : 0.2f * x; }
__device__ __forceinline__ float elu1(float x){ return x > 0.f ? x : (expf(x) - 1.f); }

// ---------------------------------------------------------------- f32 -> bf16
__global__ void cvt_bf16_kernel(const float* __restrict__ in, __bf16* __restrict__ out, long n){
  long i = (long)blockIdx.x * blockDim.x + threadIdx.x;
  long stride = (long)gridDim.x * blockDim.x;
  for (; i < n; i += stride) out[i] = (__bf16)in[i];
}

// ---------------------------------------------------------------- WMMA GEMM
// C[M,Nn] = A[M,K] * W[Nn,K]^T   (A,W bf16 row-major; C f32)
// block = 256 threads (8 waves: 2 along M x 4 along N); block tile 64x256;
// wave tile 32x64 -> 8 accumulators, 8 WMMAs per K-step.
__device__ __forceinline__ v16bf load_afrag(const __bf16* __restrict__ row, int k0, int hk) {
  // A fragment (16x32 bf16): elems 0..7 -> K = k0 + hk*8 + i ; 8..15 -> K = k0 + 16 + hk*8 + i
  v8bf lo = *(const v8bf*)(row + k0 + hk * 8);
  v8bf hi = *(const v8bf*)(row + k0 + 16 + hk * 8);
  return __builtin_shufflevector(lo, hi, 0,1,2,3,4,5,6,7,8,9,10,11,12,13,14,15);
}

__global__ __launch_bounds__(256) void wmma_gemm_bf16(
    const __bf16* __restrict__ A, const __bf16* __restrict__ W,
    float* __restrict__ C, int M, int Nn, int K)
{
  const int lane = threadIdx.x & 31;
  const int wave = threadIdx.x >> 5;
  const int wm = wave >> 2, wn = wave & 3;       // 2 x 4 wave grid
  const int nbn = Nn >> 8;                        // N tiles of 256 per row
  const int bm = blockIdx.x / nbn, bn = blockIdx.x % nbn;
  const int m0 = bm * 64 + wm * 32;
  const int n0 = bn * 256 + wn * 64;
  const int r15 = lane & 15;
  const int hk  = lane >> 4;                      // wave half (0/1)

  const __bf16* Arow0 = A + (size_t)(m0 +      r15) * K;
  const __bf16* Arow1 = A + (size_t)(m0 + 16 + r15) * K;
  const __bf16* Wr0   = W + (size_t)(n0 +  0 + r15) * K;
  const __bf16* Wr1   = W + (size_t)(n0 + 16 + r15) * K;
  const __bf16* Wr2   = W + (size_t)(n0 + 32 + r15) * K;
  const __bf16* Wr3   = W + (size_t)(n0 + 48 + r15) * K;

  v8f c00 = {}, c01 = {}, c02 = {}, c03 = {};
  v8f c10 = {}, c11 = {}, c12 = {}, c13 = {};
  for (int k0 = 0; k0 < K; k0 += 32) {
    // issue all fragment loads up front so WMMAs can use partial load waits
    v16bf a0 = load_afrag(Arow0, k0, hk);
    v16bf a1 = load_afrag(Arow1, k0, hk);
    // B fragment (32x16 bf16): lane column = lane&15, same K order as A (elems 0..7 -> k0 + hk*8 + i; 8..15 -> k0 + 16 + hk*8 + i)
    v16bf b0 = load_afrag(Wr0, k0, hk);
    v16bf b1 = load_afrag(Wr1, k0, hk);
    v16bf b2 = load_afrag(Wr2, k0, hk);
    v16bf b3 = load_afrag(Wr3, k0, hk);
    if (k0 + 64 < K) {
      __builtin_prefetch((const void*)(Arow0 + k0 + 64), 0, 3);  // near-cache (WGP) prefetch
      __builtin_prefetch((const void*)(Arow1 + k0 + 64), 0, 3);
    }
    c00 = __builtin_amdgcn_wmma_f32_16x16x32_bf16(false, a0, false, b0, (short)0, c00, false, false);
    c01 = __builtin_amdgcn_wmma_f32_16x16x32_bf16(false, a0, false, b1, (short)0, c01, false, false);
    c02 = __builtin_amdgcn_wmma_f32_16x16x32_bf16(false, a0, false, b2, (short)0, c02, false, false);
    c03 = __builtin_amdgcn_wmma_f32_16x16x32_bf16(false, a0, false, b3, (short)0, c03, false, false);
    c10 = __builtin_amdgcn_wmma_f32_16x16x32_bf16(false, a1, false, b0, (short)0, c10, false, false);
    c11 = __builtin_amdgcn_wmma_f32_16x16x32_bf16(false, a1, false, b1, (short)0, c11, false, false);
    c12 = __builtin_amdgcn_wmma_f32_16x16x32_bf16(false, a1, false, b2, (short)0, c12, false, false);
    c13 = __builtin_amdgcn_wmma_f32_16x16x32_bf16(false, a1, false, b3, (short)0, c13, false, false);
  }
  // D layout: VGPR r -> (M = r + 8*hk, N = lane&15)
  float* Cr0 = C + (size_t)(m0 +      hk * 8) * Nn + n0 + r15;
  float* Cr1 = C + (size_t)(m0 + 16 + hk * 8) * Nn + n0 + r15;
  #pragma unroll
  for (int r = 0; r < 8; r++) {
    Cr0[(size_t)r * Nn +  0] = c00[r];
    Cr0[(size_t)r * Nn + 16] = c01[r];
    Cr0[(size_t)r * Nn + 32] = c02[r];
    Cr0[(size_t)r * Nn + 48] = c03[r];
    Cr1[(size_t)r * Nn +  0] = c10[r];
    Cr1[(size_t)r * Nn + 16] = c11[r];
    Cr1[(size_t)r * Nn + 32] = c12[r];
    Cr1[(size_t)r * Nn + 48] = c13[r];
  }
}

// ---------------------------------------------------------------- attention logits
// as[n,h] = sum_c hpre[n,h*C+c]*a_src[h*C+c]; same for ad. blockDim = HC.
__global__ __launch_bounds__(1024) void attn_logits(
    const float* __restrict__ hpre, const float* __restrict__ a_src,
    const float* __restrict__ a_dst, float* __restrict__ as_o, float* __restrict__ ad_o,
    int HC, int C, int heads)
{
  __shared__ float r1[1024];
  __shared__ float r2[1024];
  const int n = blockIdx.x, t = threadIdx.x;
  float hv = hpre[(size_t)n * HC + t];
  r1[t] = hv * a_src[t];
  r2[t] = hv * a_dst[t];
  __syncthreads();
  for (int s = C >> 1; s > 0; s >>= 1) {
    if ((t & (C - 1)) < s) { r1[t] += r1[t + s]; r2[t] += r2[t + s]; }
    __syncthreads();
  }
  if ((t & (C - 1)) == 0) {
    int h = t / C;
    as_o[n * heads + h] = r1[t];
    ad_o[n * heads + h] = r2[t];
  }
}

// ---------------------------------------------------------------- CSR build
__global__ void count_edges(const int* __restrict__ ei, int E, int Etot, int* __restrict__ counts){
  int e = blockIdx.x * blockDim.x + threadIdx.x;
  if (e >= Etot) return;
  int d = (e < E) ? ei[E + e] : (e - E);
  atomicAdd(&counts[d], 1);
}

__global__ __launch_bounds__(1024) void scan_rowptr(const int* __restrict__ counts,
                                                    int* __restrict__ rowptr,
                                                    int* __restrict__ cursor, int N){
  __shared__ int lds[1024];
  const int t = threadIdx.x;
  const int per = N / 1024;             // 16 for N=16384
  int local[16];
  int sum = 0;
  for (int i = 0; i < per; i++) { int v = counts[t * per + i]; local[i] = sum; sum += v; }
  lds[t] = sum;
  __syncthreads();
  for (int s = 1; s < 1024; s <<= 1) {
    int v = (t >= s) ? lds[t - s] : 0;
    __syncthreads();
    lds[t] += v;
    __syncthreads();
  }
  int base = (t == 0) ? 0 : lds[t - 1];
  for (int i = 0; i < per; i++) {
    int r = base + local[i];
    rowptr[t * per + i] = r;
    cursor[t * per + i] = r;
  }
  if (t == 1023) rowptr[N] = lds[1023];
}

__global__ void fill_csr(const int* __restrict__ ei, int E, int Etot,
                         int* __restrict__ cursor, int* __restrict__ csr){
  int e = blockIdx.x * blockDim.x + threadIdx.x;
  if (e >= Etot) return;
  int s = (e < E) ? ei[e]     : (e - E);
  int d = (e < E) ? ei[E + e] : (e - E);
  int pos = atomicAdd(&cursor[d], 1);
  csr[pos] = s;
}

// ---------------------------------------------------------------- GAT layer-1 aggregation
// one block (256 thr) per destination node; 4 heads x 256 ch; writes bf16 (ELU'd, +b1).
__global__ __launch_bounds__(256) void gat_agg1(
    const float* __restrict__ hpre, const float* __restrict__ as1,
    const float* __restrict__ ad1, const float* __restrict__ b1,
    const int* __restrict__ rowptr, const int* __restrict__ csr,
    __bf16* __restrict__ h1b)
{
  const int d = blockIdx.x, t = threadIdx.x;
  const int base = rowptr[d], deg = rowptr[d + 1] - base;
  __shared__ float red[256];
  __shared__ float mh[4], dh[4];
  __shared__ int   sidx[64];
  __shared__ float sw[256];
  float adh[4];
  #pragma unroll
  for (int h = 0; h < 4; h++) adh[h] = ad1[d * 4 + h];

  // phase 1: per-head max
  float lm[4] = {-1e30f, -1e30f, -1e30f, -1e30f};
  for (int j = t; j < deg; j += 256) {
    int s = csr[base + j];
    #pragma unroll
    for (int h = 0; h < 4; h++) lm[h] = fmaxf(lm[h], lrelu02(as1[s * 4 + h] + adh[h]));
  }
  #pragma unroll
  for (int h = 0; h < 4; h++) {
    red[t] = lm[h]; __syncthreads();
    for (int s = 128; s > 0; s >>= 1) { if (t < s) red[t] = fmaxf(red[t], red[t + s]); __syncthreads(); }
    if (t == 0) mh[h] = red[0];
    __syncthreads();
  }
  // phase 2: per-head denom
  float ls[4] = {0, 0, 0, 0};
  for (int j = t; j < deg; j += 256) {
    int s = csr[base + j];
    #pragma unroll
    for (int h = 0; h < 4; h++) ls[h] += expf(lrelu02(as1[s * 4 + h] + adh[h]) - mh[h]);
  }
  #pragma unroll
  for (int h = 0; h < 4; h++) {
    red[t] = ls[h]; __syncthreads();
    for (int s = 128; s > 0; s >>= 1) { if (t < s) red[t] += red[t + s]; __syncthreads(); }
    if (t == 0) dh[h] = red[0] + 1e-16f;
    __syncthreads();
  }
  // phase 3: weighted gather (chunks of 64 edges)
  float acc[4] = {0, 0, 0, 0};
  for (int c0 = 0; c0 < deg; c0 += 64) {
    int cnt = min(64, deg - c0);
    if (t < cnt) sidx[t] = csr[base + c0 + t];
    __syncthreads();
    if (t < cnt * 4) {
      int j = t >> 2, h = t & 3; int s = sidx[j];
      sw[t] = expf(lrelu02(as1[s * 4 + h] + adh[h]) - mh[h]) / dh[h];
    }
    __syncthreads();
    for (int j = 0; j < cnt; j++) {
      const float* hp = hpre + (size_t)sidx[j] * 1024 + t;
      #pragma unroll
      for (int q = 0; q < 4; q++) acc[q] += sw[j * 4 + q] * hp[q * 256];
    }
    __syncthreads();
  }
  #pragma unroll
  for (int q = 0; q < 4; q++) {
    int c = q * 256 + t;
    h1b[(size_t)d * 1024 + c] = (__bf16)elu1(acc[q] + b1[c]);
  }
}

// ---------------------------------------------------------------- GAT layer-2 aggregation
// 2 heads x 128 ch; mean over heads + b2, ELU, f32 out (direct to d_out h2 region).
__global__ __launch_bounds__(256) void gat_agg2(
    const float* __restrict__ hpre, const float* __restrict__ as2,
    const float* __restrict__ ad2, const float* __restrict__ b2,
    const int* __restrict__ rowptr, const int* __restrict__ csr,
    float* __restrict__ h2out)
{
  const int d = blockIdx.x, t = threadIdx.x;
  const int base = rowptr[d], deg = rowptr[d + 1] - base;
  __shared__ float red[256];
  __shared__ float mh[2], dh[2];
  __shared__ int   sidx[64];
  __shared__ float sw[128];
  __shared__ float accs[256];
  float adh[2] = { ad2[d * 2], ad2[d * 2 + 1] };

  float lm[2] = {-1e30f, -1e30f};
  for (int j = t; j < deg; j += 256) {
    int s = csr[base + j];
    #pragma unroll
    for (int h = 0; h < 2; h++) lm[h] = fmaxf(lm[h], lrelu02(as2[s * 2 + h] + adh[h]));
  }
  #pragma unroll
  for (int h = 0; h < 2; h++) {
    red[t] = lm[h]; __syncthreads();
    for (int s = 128; s > 0; s >>= 1) { if (t < s) red[t] = fmaxf(red[t], red[t + s]); __syncthreads(); }
    if (t == 0) mh[h] = red[0];
    __syncthreads();
  }
  float ls[2] = {0, 0};
  for (int j = t; j < deg; j += 256) {
    int s = csr[base + j];
    #pragma unroll
    for (int h = 0; h < 2; h++) ls[h] += expf(lrelu02(as2[s * 2 + h] + adh[h]) - mh[h]);
  }
  #pragma unroll
  for (int h = 0; h < 2; h++) {
    red[t] = ls[h]; __syncthreads();
    for (int s = 128; s > 0; s >>= 1) { if (t < s) red[t] += red[t + s]; __syncthreads(); }
    if (t == 0) dh[h] = red[0] + 1e-16f;
    __syncthreads();
  }
  float acc = 0.f;
  const int myh = t >> 7;
  for (int c0 = 0; c0 < deg; c0 += 64) {
    int cnt = min(64, deg - c0);
    if (t < cnt) sidx[t] = csr[base + c0 + t];
    __syncthreads();
    if (t < cnt * 2) {
      int j = t >> 1, h = t & 1; int s = sidx[j];
      sw[t] = expf(lrelu02(as2[s * 2 + h] + adh[h]) - mh[h]) / dh[h];
    }
    __syncthreads();
    for (int j = 0; j < cnt; j++)
      acc += sw[j * 2 + myh] * hpre[(size_t)sidx[j] * 256 + t];
    __syncthreads();
  }
  accs[t] = acc;
  __syncthreads();
  if (t < 128) {
    float v = 0.5f * (accs[t] + accs[t + 128]) + b2[t];
    h2out[(size_t)d * 128 + t] = elu1(v);
  }
}

// ---------------------------------------------------------------- pooling helpers
__global__ void graph_counts(const int* __restrict__ batch, int n, float* __restrict__ ng){
  __shared__ float h[NGRAPH];
  int t = threadIdx.x;
  if (t < NGRAPH) h[t] = 0.f;
  __syncthreads();
  int i = blockIdx.x * blockDim.x + t;
  if (i < n) atomicAdd(&h[batch[i]], 1.0f);
  __syncthreads();
  if (t < NGRAPH) atomicAdd(&ng[t], h[t]);
}

__global__ void edge_counts(const int* __restrict__ ei, const int* __restrict__ batch,
                            int E, float* __restrict__ eg){
  __shared__ float h[NGRAPH];
  int t = threadIdx.x;
  if (t < NGRAPH) h[t] = 0.f;
  __syncthreads();
  int e = blockIdx.x * blockDim.x + t;
  if (e < E) atomicAdd(&h[batch[ei[e]]], 1.0f);
  __syncthreads();
  if (t < NGRAPH) atomicAdd(&eg[t], h[t]);
}

__global__ __launch_bounds__(128) void pool_sum(const float* __restrict__ h2,
                                                const int* __restrict__ batch,
                                                float* __restrict__ pooled){
  const int t = threadIdx.x;
  const int n0 = blockIdx.x * 128;
  float acc = 0.f;
  int cur = batch[n0];
  for (int i = 0; i < 128; i++) {
    int n = n0 + i;
    int bb = batch[n];
    float v = h2[(size_t)n * OUTC + t];
    if (bb != cur) { atomicAdd(&pooled[cur * OUTC + t], acc); acc = 0.f; cur = bb; }
    acc += v;
  }
  atomicAdd(&pooled[cur * OUTC + t], acc);
}

// ---------------------------------------------------------------- head: proj + LN + risk MLP
__global__ __launch_bounds__(768) void head_kernel(
    const float* __restrict__ pooledSum, const float* __restrict__ ng,
    const float* __restrict__ eg, const float* __restrict__ projW,
    const float* __restrict__ projB, const float* __restrict__ lng,
    const float* __restrict__ lnb, const float* __restrict__ rhW1,
    const float* __restrict__ rhb1, const float* __restrict__ rhW2,
    const float* __restrict__ rhb2, float* __restrict__ out_se,
    float* __restrict__ out_risk)
{
  __shared__ float sp[NGRAPH * OUTC];     // mean-pooled
  __shared__ float red[768];
  __shared__ float stats[NGRAPH * 2];
  __shared__ float ri[NGRAPH * (OUTC + 2)];
  __shared__ float hidv[NGRAPH * 32];
  const int t = threadIdx.x;

  for (int i = t; i < NGRAPH * OUTC; i += 768) {
    int b = i >> 7;
    sp[i] = pooledSum[i] / fmaxf(ng[b], 1.0f);
  }
  if (t < NGRAPH * 2) {
    int b = t >> 1;
    stats[t] = (t & 1) ? logf(ng[b] + 1.0f) : eg[b] / (ng[b] + 1e-6f);
  }
  __syncthreads();

  // pe[b][t] = projB[t] + sum_c sp[b][c]*projW[t][c]
  float pe[NGRAPH];
  for (int b = 0; b < NGRAPH; b++) {
    float s = projB[t];
    const float* w = projW + (size_t)t * OUTC;
    const float* p = sp + b * OUTC;
    for (int c = 0; c < OUTC; c++) s += w[c] * p[c];
    pe[b] = s;
  }
  // layernorm per graph over 768 features
  for (int b = 0; b < NGRAPH; b++) {
    red[t] = pe[b]; __syncthreads();
    for (int s = 512; s > 0; s >>= 1) { if (t < s && t + s < 768) red[t] += red[t + s]; __syncthreads(); }
    float mu = red[0] / 768.0f;
    __syncthreads();
    float dv = pe[b] - mu;
    red[t] = dv * dv; __syncthreads();
    for (int s = 512; s > 0; s >>= 1) { if (t < s && t + s < 768) red[t] += red[t + s]; __syncthreads(); }
    float var = red[0] / 768.0f;
    __syncthreads();
    out_se[b * 768 + t] = dv * rsqrtf(var + 1e-5f) * lng[t] + lnb[t];
  }
  // risk head: ri = [pooled, stats] -> 32 relu -> 1 sigmoid
  for (int i = t; i < NGRAPH * (OUTC + 2); i += 768) {
    int b = i / (OUTC + 2), j = i % (OUTC + 2);
    ri[i] = (j < OUTC) ? sp[b * OUTC + j] : stats[b * 2 + (j - OUTC)];
  }
  __syncthreads();
  if (t < NGRAPH * 32) {
    int b = t >> 5, u = t & 31;
    float s = rhb1[u];
    const float* w = rhW1 + u * (OUTC + 2);
    const float* r = ri + b * (OUTC + 2);
    for (int i = 0; i < OUTC + 2; i++) s += w[i] * r[i];
    hidv[b * 32 + u] = fmaxf(s, 0.f);
  }
  __syncthreads();
  if (t < NGRAPH) {
    float s = rhb2[0];
    for (int u = 0; u < 32; u++) s += hidv[t * 32 + u] * rhW2[u];
    out_risk[t] = 1.0f / (1.0f + expf(-s));
  }
}

// ---------------------------------------------------------------- launch
extern "C" void kernel_launch(void* const* d_in, const int* in_sizes, int n_in,
                              void* d_out, int out_size, void* d_ws, size_t ws_size,
                              hipStream_t stream)
{
  const float* x      = (const float*)d_in[0];
  const int*   ei     = (const int*)  d_in[1];
  const int*   batch  = (const int*)  d_in[2];
  const float* W1     = (const float*)d_in[3];
  const float* a_src1 = (const float*)d_in[4];
  const float* a_dst1 = (const float*)d_in[5];
  const float* b1     = (const float*)d_in[6];
  const float* W2     = (const float*)d_in[7];
  const float* a_src2 = (const float*)d_in[8];
  const float* a_dst2 = (const float*)d_in[9];
  const float* b2     = (const float*)d_in[10];
  const float* projW  = (const float*)d_in[11];
  const float* projB  = (const float*)d_in[12];
  const float* lng    = (const float*)d_in[13];
  const float* lnb    = (const float*)d_in[14];
  const float* rhW1   = (const float*)d_in[15];
  const float* rhb1   = (const float*)d_in[16];
  const float* rhW2   = (const float*)d_in[17];
  const float* rhb2   = (const float*)d_in[18];

  const int N    = in_sizes[0] / INDIM;   // 16384
  const int E    = in_sizes[1] / 2;       // 262144
  const int Etot = E + N;                 // with self-loops
  const int HC1  = H1N * HID1;            // 1024
  const int HC2  = H2N * OUTC;            // 256

  char* ws = (char*)d_ws;
  size_t off = 0;
  auto alloc = [&](size_t bytes) -> char* {
    char* p = ws + off;
    off = (off + bytes + 255) & ~(size_t)255;
    return p;
  };
  __bf16* xb     = (__bf16*)alloc((size_t)N * INDIM * 2);
  __bf16* w1b    = (__bf16*)alloc((size_t)HC1 * INDIM * 2);
  __bf16* w2b    = (__bf16*)alloc((size_t)HC2 * HC1 * 2);
  float*  h1pre  = (float*) alloc((size_t)N * HC1 * 4);
  __bf16* h1b    = (__bf16*)alloc((size_t)N * HC1 * 2);
  float*  h2pre  = (float*) alloc((size_t)N * HC2 * 4);
  float*  as1p   = (float*) alloc((size_t)N * H1N * 4);
  float*  ad1p   = (float*) alloc((size_t)N * H1N * 4);
  float*  as2p   = (float*) alloc((size_t)N * H2N * 4);
  float*  ad2p   = (float*) alloc((size_t)N * H2N * 4);
  int*    counts = (int*)   alloc((size_t)N * 4);
  int*    rowptr = (int*)   alloc((size_t)(N + 1) * 4);
  int*    cursor = (int*)   alloc((size_t)N * 4);
  int*    csr    = (int*)   alloc((size_t)Etot * 4);
  float*  pooled = (float*) alloc((size_t)NGRAPH * OUTC * 4);
  float*  ngp    = (float*) alloc(NGRAPH * 4);
  float*  egp    = (float*) alloc(NGRAPH * 4);

  (void)hipMemsetAsync(counts, 0, (size_t)N * 4, stream);
  (void)hipMemsetAsync(pooled, 0, (size_t)NGRAPH * OUTC * 4, stream);
  (void)hipMemsetAsync(ngp, 0, NGRAPH * 4, stream);
  (void)hipMemsetAsync(egp, 0, NGRAPH * 4, stream);

  // bf16 packing
  cvt_bf16_kernel<<<512, 256, 0, stream>>>(x,  xb,  (long)N * INDIM);
  cvt_bf16_kernel<<<256, 256, 0, stream>>>(W1, w1b, (long)HC1 * INDIM);
  cvt_bf16_kernel<<<256, 256, 0, stream>>>(W2, w2b, (long)HC2 * HC1);

  // CSR by destination (incl. self-loops)
  count_edges<<<(Etot + 255) / 256, 256, 0, stream>>>(ei, E, Etot, counts);
  scan_rowptr<<<1, 1024, 0, stream>>>(counts, rowptr, cursor, N);
  fill_csr<<<(Etot + 255) / 256, 256, 0, stream>>>(ei, E, Etot, cursor, csr);

  // layer 1
  wmma_gemm_bf16<<<(N / 64) * (HC1 / 256), 256, 0, stream>>>(xb, w1b, h1pre, N, HC1, INDIM);
  attn_logits<<<N, HC1, 0, stream>>>(h1pre, a_src1, a_dst1, as1p, ad1p, HC1, HID1, H1N);
  gat_agg1<<<N, 256, 0, stream>>>(h1pre, as1p, ad1p, b1, rowptr, csr, h1b);

  // layer 2
  float* out_se   = (float*)d_out;
  float* out_h2   = out_se + NGRAPH * INDIM;
  float* out_risk = out_h2 + (size_t)N * OUTC;
  wmma_gemm_bf16<<<(N / 64) * (HC2 / 256), 256, 0, stream>>>(h1b, w2b, h2pre, N, HC2, HC1);
  attn_logits<<<N, HC2, 0, stream>>>(h2pre, a_src2, a_dst2, as2p, ad2p, HC2, OUTC, H2N);
  gat_agg2<<<N, 256, 0, stream>>>(h2pre, as2p, ad2p, b2, rowptr, csr, out_h2);

  // pooling + head
  graph_counts<<<(N + 255) / 256, 256, 0, stream>>>(batch, N, ngp);
  edge_counts<<<(E + 255) / 256, 256, 0, stream>>>(ei, batch, E, egp);
  pool_sum<<<N / 128, 128, 0, stream>>>(out_h2, batch, pooled);
  head_kernel<<<1, 768, 0, stream>>>(pooled, ngp, egp, projW, projB, lng, lnb,
                                     rhW1, rhb1, rhW2, rhb2, out_se, out_risk);
}
